// GraphResNetBlock_53532472377793
// MI455X (gfx1250) — matrix operand order repaired and checked by reading, not compiled, not run on hardware
//
#include <hip/hip_runtime.h>

// ---------------- problem constants ----------------
#define BB   2
#define VV   65536
#define CIN  128
#define COUT 256
#define TDIM 512
#define G1N  32
#define G2N  32
#define EPSF 1e-5f

#define K1   (9*CIN)    // 1152
#define K2   (9*COUT)   // 2304

// LDS row strides (halves): pad for bank spread, keep rows 16B-aligned
#define LDA1 (K1+8)     // 1160 halves = 2320 B (16 | 2320)
#define LDA2 (K2+8)     // 2312 halves = 4624 B (16 | 4624)
#define LDSX (CIN+8)    // 136

typedef __attribute__((ext_vector_type(16))) _Float16 v16h;
typedef __attribute__((ext_vector_type(8)))  _Float16 v8h;
typedef __attribute__((ext_vector_type(4)))  _Float16 v4h;
typedef __attribute__((ext_vector_type(8)))  float    v8f;
typedef __attribute__((ext_vector_type(8)))  unsigned int u32x8;
typedef __attribute__((ext_vector_type(4)))  unsigned int u32x4;
typedef __attribute__((ext_vector_type(2)))  unsigned int u32x2;

__device__ __forceinline__ float siluf(float v) {
    return v * (1.0f / (1.0f + __expf(-v)));
}

// A-fragment (16x32 f16, same k order as ldB): lane (l,hi) = row l,
// halves K = k0 + hi*16 + {0..15}
__device__ __forceinline__ v16h ldsA(const _Float16* row, int k0, int hi) {
    const _Float16* p0 = row + k0 + hi * 16;
    const _Float16* p1 = p0 + 8;
    u32x2 a = *(const u32x2*)p0;
    u32x2 b = *(const u32x2*)(p0 + 4);
    u32x2 c = *(const u32x2*)p1;
    u32x2 d = *(const u32x2*)(p1 + 4);
    u32x8 r;
    r[0]=a[0]; r[1]=a[1]; r[2]=b[0]; r[3]=b[1];
    r[4]=c[0]; r[5]=c[1]; r[6]=d[0]; r[7]=d[1];
    return __builtin_bit_cast(v16h, r);
}

// B-fragment (32x16 f16, column-striped): lane (l,hi) = col n0+l,
// halves K = k0 + hi*16 + {0..15}; W rows are K-major -> 32 contiguous bytes.
__device__ __forceinline__ v16h ldB(const _Float16* p) {
    u32x4 a = *(const u32x4*)p;
    u32x4 b = *(const u32x4*)(p + 8);
    u32x8 r;
    r[0]=a[0]; r[1]=a[1]; r[2]=a[2]; r[3]=a[3];
    r[4]=b[0]; r[5]=b[1]; r[6]=b[2]; r[7]=b[3];
    return __builtin_bit_cast(v16h, r);
}

__device__ __forceinline__ v8f wmma16(v16h a, v16h b, v8f c) {
    return __builtin_amdgcn_wmma_f32_16x16x32_f16(false, a, false, b,
                                                  (short)0, c, false, false);
}

// Async global->LDS 16B copy (per-lane gather), tracked by ASYNCcnt.
// LDS operand = low 32 bits of the generic pointer (== LDS offset on AMDGPU).
__device__ __forceinline__ void async_g2l_b128(const void* gptr, const void* lptr) {
    unsigned lo = (unsigned)(size_t)lptr;
    asm volatile("global_load_async_to_lds_b128 %0, %1, off"
                 :: "v"(lo), "v"(gptr) : "memory");
}
__device__ __forceinline__ void wait_async0() {
#if __has_builtin(__builtin_amdgcn_s_wait_asynccnt)
    __builtin_amdgcn_s_wait_asynccnt(0);
#else
    asm volatile("s_wait_asynccnt 0x0" ::: "memory");
#endif
}

// ---------------- kernel 1: fp32 -> fp16 weights ----------------
#define NW1 (COUT*K1)        // 294912
#define NW2 (COUT*K2)        // 589824
#define NWS (COUT*CIN)       // 32768
__global__ void cvt_weights(const float* __restrict__ w1, const float* __restrict__ w2,
                            const float* __restrict__ wsm,
                            _Float16* __restrict__ o1, _Float16* __restrict__ o2,
                            _Float16* __restrict__ o3) {
    const int total = NW1 + NW2 + NWS;
    for (int i = blockIdx.x * blockDim.x + threadIdx.x; i < total;
         i += gridDim.x * blockDim.x) {
        if (i < NW1)            o1[i]            = (_Float16)w1[i];
        else if (i < NW1 + NW2) o2[i - NW1]       = (_Float16)w2[i - NW1];
        else                    o3[i - NW1 - NW2] = (_Float16)wsm[i - NW1 - NW2];
    }
}

// ---------------- GN1 stats: partial sums per (b,g,chunk) ----------------
__global__ void gn1_partial(const float* __restrict__ x, float2* __restrict__ part) {
    const int blk = blockIdx.x;
    const int chunk = blk & 15;
    const int gidx  = blk >> 4;          // b*32+g
    const int b = gidx >> 5, g = gidx & 31;
    const float* xp = x + (size_t)b * VV * CIN + g * 4;
    const int vstart = chunk * (VV / 16);
    float s = 0.f, q = 0.f;
    for (int i = threadIdx.x; i < VV / 16; i += 256) {
        float4 v = *(const float4*)(xp + (size_t)(vstart + i) * CIN);
        s += v.x + v.y + v.z + v.w;
        q += v.x*v.x + v.y*v.y + v.z*v.z + v.w*v.w;
    }
    __shared__ float rs[256], rq[256];
    rs[threadIdx.x] = s; rq[threadIdx.x] = q;
    __syncthreads();
    for (int o = 128; o > 0; o >>= 1) {
        if (threadIdx.x < o) { rs[threadIdx.x] += rs[threadIdx.x+o];
                               rq[threadIdx.x] += rq[threadIdx.x+o]; }
        __syncthreads();
    }
    if (threadIdx.x == 0) { float2 p; p.x = rs[0]; p.y = rq[0]; part[blk] = p; }
}

__global__ void gn_finalize(const float2* __restrict__ part, float2* __restrict__ st,
                            float invN) {
    const int i = threadIdx.x;
    float s = 0.f, q = 0.f;
    for (int c = 0; c < 16; ++c) { float2 p = part[i*16 + c]; s += p.x; q += p.y; }
    float mean = s * invN;
    float var  = q * invN - mean * mean;
    float2 o; o.x = mean; o.y = rsqrtf(var + EPSF);
    st[i] = o;
}

// ---------------- h1 = f16(silu(gn1(x))) ----------------
__global__ void h1_kernel(const float* __restrict__ x, const float2* __restrict__ st1,
                          const float* __restrict__ gw, const float* __restrict__ gb,
                          _Float16* __restrict__ h1) {
    const int idx = blockIdx.x * 256 + threadIdx.x;
    const int e0 = idx * 4;
    const int c0 = e0 & (CIN - 1);
    const int b  = e0 >> 23;               // V*CIN = 2^23
    float2 st = st1[b * G1N + (c0 >> 2)];  // 4 channels per group
    float4 xv = *(const float4*)(x + (size_t)e0);
    float vals[4] = { xv.x, xv.y, xv.z, xv.w };
    v4h hv;
    #pragma unroll
    for (int e = 0; e < 4; ++e) {
        float f = (vals[e] - st.x) * st.y * gw[c0 + e] + gb[c0 + e];
        hv[e] = (_Float16)siluf(f);
    }
    *(v4h*)(h1 + e0) = hv;
}

// ---------------- tvec = silu(t_emb) @ wt^T + bt ----------------
__global__ void tvec_kernel(const float* __restrict__ t_emb, const float* __restrict__ wt,
                            const float* __restrict__ bt, float* __restrict__ tvec) {
    const int b = blockIdx.x, o = threadIdx.x;
    const float* te = t_emb + (size_t)b * TDIM;
    const float* wr = wt + (size_t)o * TDIM;
    float acc = bt[o];
    for (int k = 0; k < TDIM; ++k) acc += siluf(te[k]) * wr[k];
    tvec[b * COUT + o] = acc;
}

// ---------------- conv1: hmid = gather(h1)@w1^T + b1 + tvec ----------------
// M-tile = 32 vertices/block; 8 waves = 2 M-halves x 4 N-groups (4 tiles each).
__global__ void __launch_bounds__(256)
conv1_kernel(const _Float16* __restrict__ h1, const _Float16* __restrict__ w1h,
             const int* __restrict__ neigh, const float* __restrict__ b1,
             const float* __restrict__ tvec, _Float16* __restrict__ hmid) {
    __shared__ _Float16 As[32 * LDA1];      // 74.2 KB
    __shared__ int sN[288];
    const int tid = threadIdx.x;
    const int blk = blockIdx.x;
    const int b  = blk >> 11;               // V/32 = 2048 tiles per batch
    const int v0 = (blk & 2047) << 5;

    for (int i = tid; i < 288; i += 256)
        sN[i] = neigh[(v0 + i / 9) * 9 + i % 9];
    __syncthreads();

    // stage A via async global->LDS gather: 32 rows x 9 segs x 16 b128 units
    const _Float16* h1b = h1 + (size_t)b * VV * CIN;
    for (int u = tid; u < 32 * 9 * 16; u += 256) {
        const int r = u / 144, rem = u - r * 144;
        const int n = rem >> 4, j = rem & 15;
        const int nv = sN[r * 9 + n];
        async_g2l_b128(h1b + (size_t)nv * CIN + j * 8,
                       As + r * LDA1 + n * CIN + j * 8);
    }
    wait_async0();
    __syncthreads();

    const int lane = tid & 31, wave = tid >> 5;
    const int l = lane & 15, hi = lane >> 4;
    const int mh = wave >> 2;               // rows mh*16 .. mh*16+15
    const int ng = wave & 3;                // N tiles ng*64 + t*16
    const _Float16* Arow = As + (mh * 16 + l) * LDA1;
    const _Float16* wp[4];
    #pragma unroll
    for (int t = 0; t < 4; ++t)
        wp[t] = w1h + (size_t)(ng * 64 + t * 16 + l) * K1 + hi * 16;

    v8f acc[4] = {};
    for (int k0 = 0; k0 < K1; k0 += 32) {
        __builtin_prefetch(wp[0] + k0 + 256, 0, 3);
        v16h a = ldsA(Arow, k0, hi);
        #pragma unroll
        for (int t = 0; t < 4; ++t) acc[t] = wmma16(a, ldB(wp[t] + k0), acc[t]);
    }

    _Float16* outb = hmid + (size_t)b * VV * COUT + (size_t)(v0 + mh * 16) * COUT;
    const float* tv = tvec + b * COUT;
    #pragma unroll
    for (int t = 0; t < 4; ++t) {
        const int o = ng * 64 + t * 16 + l;
        const float add = b1[o] + tv[o];
        #pragma unroll
        for (int r = 0; r < 8; ++r)
            outb[(size_t)(hi * 8 + r) * COUT + o] = (_Float16)(acc[t][r] + add);
    }
}

// ---------------- GN2 stats over hmid (f16) ----------------
__global__ void gn2_partial(const _Float16* __restrict__ hmid, float2* __restrict__ part) {
    const int blk = blockIdx.x;
    const int chunk = blk & 15;
    const int gidx  = blk >> 4;
    const int b = gidx >> 5, g = gidx & 31;
    const _Float16* hp = hmid + (size_t)b * VV * COUT + g * 8;   // 8 ch / group
    const int vstart = chunk * (VV / 16);
    float s = 0.f, q = 0.f;
    for (int i = threadIdx.x; i < VV / 16; i += 256) {
        u32x4 d = *(const u32x4*)(hp + (size_t)(vstart + i) * COUT);
        v8h hv = __builtin_bit_cast(v8h, d);
        #pragma unroll
        for (int e = 0; e < 8; ++e) { float f = (float)hv[e]; s += f; q += f * f; }
    }
    __shared__ float rs[256], rq[256];
    rs[threadIdx.x] = s; rq[threadIdx.x] = q;
    __syncthreads();
    for (int o = 128; o > 0; o >>= 1) {
        if (threadIdx.x < o) { rs[threadIdx.x] += rs[threadIdx.x+o];
                               rq[threadIdx.x] += rq[threadIdx.x+o]; }
        __syncthreads();
    }
    if (threadIdx.x == 0) { float2 p; p.x = rs[0]; p.y = rq[0]; part[blk] = p; }
}

// ---------------- conv2 + shortcut -> out (f32) ----------------
// M-tile = 32; A-staging applies GN2+SiLU on the fly; shortcut GEMM fused.
__global__ void __launch_bounds__(256)
conv2_kernel(const _Float16* __restrict__ hmid, const float* __restrict__ x,
             const _Float16* __restrict__ w2h, const _Float16* __restrict__ wsh,
             const int* __restrict__ neigh, const float2* __restrict__ st2,
             const float* __restrict__ g2w, const float* __restrict__ g2b,
             const float* __restrict__ b2, const float* __restrict__ bs,
             float* __restrict__ out) {
    __shared__ _Float16 As[32 * LDA2];      // 148 KB
    __shared__ _Float16 Sx[32 * LDSX];      // 8.7 KB
    __shared__ int sN[288];
    const int tid = threadIdx.x;
    const int blk = blockIdx.x;
    const int b  = blk >> 11;
    const int v0 = (blk & 2047) << 5;

    for (int i = tid; i < 288; i += 256)
        sN[i] = neigh[(v0 + i / 9) * 9 + i % 9];
    __syncthreads();

    // stage A with fused GN2+SiLU: unit = 8 halves = one 8-channel group
    const _Float16* hb = hmid + (size_t)b * VV * COUT;
    for (int u = tid; u < 32 * 9 * 32; u += 256) {
        const int r = u / 288, rem = u - r * 288;
        const int n = rem >> 5, j = rem & 31;   // j == group id (8 ch each)
        const int nv = sN[r * 9 + n];
        const int c0 = j * 8;
        const float2 st = st2[b * G2N + j];
        u32x4 d = *(const u32x4*)(hb + (size_t)nv * COUT + c0);
        v8h hv = __builtin_bit_cast(v8h, d);
        v8h ov;
        #pragma unroll
        for (int e = 0; e < 8; ++e) {
            float f = ((float)hv[e] - st.x) * st.y * g2w[c0 + e] + g2b[c0 + e];
            ov[e] = (_Float16)siluf(f);
        }
        *(u32x4*)(As + r * LDA2 + n * COUT + c0) = __builtin_bit_cast(u32x4, ov);
    }
    // stage shortcut tile: x (f32) -> f16
    const float* xb = x + (size_t)b * VV * CIN;
    for (int u = tid; u < 32 * 32; u += 256) {
        const int r = u >> 5, j = u & 31;
        float4 xv = *(const float4*)(xb + (size_t)(v0 + r) * CIN + j * 4);
        v4h hv;
        hv[0] = (_Float16)xv.x; hv[1] = (_Float16)xv.y;
        hv[2] = (_Float16)xv.z; hv[3] = (_Float16)xv.w;
        *(v4h*)(Sx + r * LDSX + j * 4) = hv;
    }
    __syncthreads();

    const int lane = tid & 31, wave = tid >> 5;
    const int l = lane & 15, hi = lane >> 4;
    const int mh = wave >> 2;
    const int ng = wave & 3;
    const _Float16* Arow = As + (mh * 16 + l) * LDA2;
    const _Float16* Xrow = Sx + (mh * 16 + l) * LDSX;
    const _Float16* wp[4];
    const _Float16* sp[4];
    #pragma unroll
    for (int t = 0; t < 4; ++t) {
        const int o = ng * 64 + t * 16 + l;
        wp[t] = w2h + (size_t)o * K2  + hi * 16;
        sp[t] = wsh + (size_t)o * CIN + hi * 16;
    }

    v8f acc[4] = {};
    for (int k0 = 0; k0 < K2; k0 += 32) {
        __builtin_prefetch(wp[0] + k0 + 256, 0, 3);
        v16h a = ldsA(Arow, k0, hi);
        #pragma unroll
        for (int t = 0; t < 4; ++t) acc[t] = wmma16(a, ldB(wp[t] + k0), acc[t]);
    }
    for (int k0 = 0; k0 < CIN; k0 += 32) {     // fused shortcut GEMM
        v16h a = ldsA(Xrow, k0, hi);
        #pragma unroll
        for (int t = 0; t < 4; ++t) acc[t] = wmma16(a, ldB(sp[t] + k0), acc[t]);
    }

    float* outb = out + (size_t)b * VV * COUT + (size_t)(v0 + mh * 16) * COUT;
    #pragma unroll
    for (int t = 0; t < 4; ++t) {
        const int o = ng * 64 + t * 16 + l;
        const float add = b2[o] + bs[o];
        #pragma unroll
        for (int r = 0; r < 8; ++r)
            outb[(size_t)(hi * 8 + r) * COUT + o] = acc[t][r] + add;
    }
}

// ---------------- workspace layout (bytes) ----------------
static const size_t OFF_H1    = 0;                                  // 32 MB f16
static const size_t OFF_HMID  = OFF_H1   + (size_t)BB*VV*CIN*2;     // 64 MB f16
static const size_t OFF_W1H   = OFF_HMID + (size_t)BB*VV*COUT*2;
static const size_t OFF_W2H   = OFF_W1H  + (size_t)NW1*2;
static const size_t OFF_WSH   = OFF_W2H  + (size_t)NW2*2;
static const size_t OFF_TVEC  = OFF_WSH  + (size_t)NWS*2;
static const size_t OFF_PART1 = OFF_TVEC  + (size_t)BB*COUT*4;
static const size_t OFF_ST1   = OFF_PART1 + 1024*sizeof(float2);
static const size_t OFF_PART2 = OFF_ST1   + 64*sizeof(float2);
static const size_t OFF_ST2   = OFF_PART2 + 1024*sizeof(float2);

extern "C" void kernel_launch(void* const* d_in, const int* in_sizes, int n_in,
                              void* d_out, int out_size, void* d_ws, size_t ws_size,
                              hipStream_t stream) {
    const float* x     = (const float*)d_in[0];
    const float* t_emb = (const float*)d_in[1];
    const int*   neigh = (const int*)  d_in[2];
    const float* gn1_w = (const float*)d_in[3];
    const float* gn1_b = (const float*)d_in[4];
    const float* w1    = (const float*)d_in[5];
    const float* b1    = (const float*)d_in[6];
    const float* wt    = (const float*)d_in[7];
    const float* bt    = (const float*)d_in[8];
    const float* gn2_w = (const float*)d_in[9];
    const float* gn2_b = (const float*)d_in[10];
    const float* w2    = (const float*)d_in[11];
    const float* b2    = (const float*)d_in[12];
    const float* wsm   = (const float*)d_in[13];
    const float* bs    = (const float*)d_in[14];
    float* out = (float*)d_out;

    char* wsb = (char*)d_ws;
    _Float16* h1   = (_Float16*)(wsb + OFF_H1);
    _Float16* hmid = (_Float16*)(wsb + OFF_HMID);
    _Float16* w1h  = (_Float16*)(wsb + OFF_W1H);
    _Float16* w2h  = (_Float16*)(wsb + OFF_W2H);
    _Float16* wsh  = (_Float16*)(wsb + OFF_WSH);
    float*    tvec = (float*)   (wsb + OFF_TVEC);
    float2*   part1= (float2*)  (wsb + OFF_PART1);
    float2*   st1  = (float2*)  (wsb + OFF_ST1);
    float2*   part2= (float2*)  (wsb + OFF_PART2);
    float2*   st2  = (float2*)  (wsb + OFF_ST2);

    cvt_weights<<<1024, 256, 0, stream>>>(w1, w2, wsm, w1h, w2h, wsh);

    gn1_partial<<<BB*G1N*16, 256, 0, stream>>>(x, part1);
    gn_finalize<<<1, 64, 0, stream>>>(part1, st1, 1.0f / (float)(VV * (CIN / G1N)));
    h1_kernel<<<(BB*VV*CIN/4)/256, 256, 0, stream>>>(x, st1, gn1_w, gn1_b, h1);

    tvec_kernel<<<BB, COUT, 0, stream>>>(t_emb, wt, bt, tvec);

    conv1_kernel<<<BB*VV/32, 256, 0, stream>>>(h1, w1h, neigh, b1, tvec, hmid);

    gn2_partial<<<BB*G2N*16, 256, 0, stream>>>(hmid, part2);
    gn_finalize<<<1, 64, 0, stream>>>(part2, st2, 1.0f / (float)(VV * (COUT / G2N)));

    conv2_kernel<<<BB*VV/32, 256, 0, stream>>>(hmid, x, w2h, wsh, neigh, st2,
                                               gn2_w, gn2_b, b2, bs, out);
    (void)in_sizes; (void)n_in; (void)out_size; (void)ws_size;
}
